// T2T_ViT_57140244906021
// MI455X (gfx1250) — compile-verified, matrix operand order checked by reading, not run on hardware
//
#include <hip/hip_runtime.h>
#include <hip/hip_bf16.h>
#include <math.h>

typedef _Float16 h16;
typedef _Float16 v8h  __attribute__((ext_vector_type(8)));
typedef _Float16 v16h __attribute__((ext_vector_type(16)));
typedef float    v8f  __attribute__((ext_vector_type(8)));
typedef int      v4i  __attribute__((ext_vector_type(4)));

#if __has_builtin(__builtin_amdgcn_global_load_async_to_lds_b128) && \
    __has_builtin(__builtin_amdgcn_s_wait_asynccnt)
#define HAVE_ASYNC_LDS 1
#else
#define HAVE_ASYNC_LDS 0
#endif

// ---------------------------------------------------------------- helpers

__device__ __forceinline__ v16h ld_frag(const h16* p0, const h16* p1) {
  v8h lo = *(const v8h*)p0;
  v8h hi = *(const v8h*)p1;
  return __builtin_shufflevector(lo, hi, 0,1,2,3,4,5,6,7,8,9,10,11,12,13,14,15);
}

__device__ __forceinline__ v8f wmma32(v16h a, v16h b, v8f c) {
  // (neg_a, A, neg_b, B, c_mod, C, reuse_a, reuse_b)
  return __builtin_amdgcn_wmma_f32_16x16x32_f16(false, a, false, b, (short)0, c,
                                                false, false);
}

__device__ __forceinline__ float wave_sum(float v) {
#pragma unroll
  for (int m = 16; m >= 1; m >>= 1) v += __shfl_xor(v, m, 32);
  return v;
}
__device__ __forceinline__ float wave_max(float v) {
#pragma unroll
  for (int m = 16; m >= 1; m >>= 1) v = fmaxf(v, __shfl_xor(v, m, 32));
  return v;
}

// stage 16 bytes global -> LDS (async on CDNA5, sync fallback otherwise)
__device__ __forceinline__ void stage16(const h16* src, h16* dst_lds) {
#if HAVE_ASYNC_LDS
  typedef __attribute__((address_space(1))) v4i* gv4i_p;
  typedef __attribute__((address_space(3))) v4i* lv4i_p;
  __builtin_amdgcn_global_load_async_to_lds_b128((gv4i_p)(void*)(h16*)src,
                                                 (lv4i_p)dst_lds, 0, 0);
#else
  *(v8h*)dst_lds = *(const v8h*)src;
#endif
}
__device__ __forceinline__ void stage_wait_one() {
#if HAVE_ASYNC_LDS
  __builtin_amdgcn_s_wait_asynccnt(1);
#endif
}
__device__ __forceinline__ void stage_wait_zero() {
#if HAVE_ASYNC_LDS
  __builtin_amdgcn_s_wait_asynccnt(0);
#endif
}

// ---------------------------------------------------------------- embed

__global__ __launch_bounds__(256) void embed_k(const float* __restrict__ x,
                                               const float* __restrict__ cls,
                                               float* __restrict__ h) {
  long long idx = (long long)blockIdx.x * blockDim.x + threadIdx.x;
  const long long total = 64LL * 197 * 768;
  if (idx >= total) return;
  int c = (int)(idx % 768);
  long long t = idx / 768;
  int n = (int)(t % 197);
  int b = (int)(t / 197);
  float base = (n == 0) ? cls[c] : x[((long long)b * 196 + (n - 1)) * 768 + c];
  int i2 = c & ~1;
  float dv = __expf(-9.210340371976184f * (float)i2 * (1.0f / 768.0f));
  float ang = (float)n * dv;
  float pe = (c & 1) ? __cosf(ang) : __sinf(ang);
  h[idx] = base + pe;
}

// ---------------------------------------------------------------- f32 -> f16

__global__ __launch_bounds__(256) void cvt_f16_k(const float* __restrict__ in,
                                                 h16* __restrict__ out,
                                                 long long n) {
  long long i = (long long)blockIdx.x * blockDim.x + threadIdx.x;
  long long stride = (long long)gridDim.x * blockDim.x;
  for (; i < n; i += stride) out[i] = (h16)in[i];
}

// ---------------------------------------------------------------- layernorm

__global__ __launch_bounds__(256) void layernorm_k(const float* __restrict__ x,
                                                   const float* __restrict__ w,
                                                   const float* __restrict__ bvec,
                                                   h16* __restrict__ y,
                                                   int rows, int C) {
  int row  = blockIdx.x * 8 + (threadIdx.x >> 5);
  int lane = threadIdx.x & 31;
  if (row >= rows) return;
  const float* xp = x + (long long)row * C;
  float s1 = 0.f, s2 = 0.f;
  for (int c = lane; c < C; c += 32) { float v = xp[c]; s1 += v; s2 += v * v; }
  s1 = wave_sum(s1);
  s2 = wave_sum(s2);
  float mu  = s1 / (float)C;
  float var = s2 / (float)C - mu * mu;
  float inv = rsqrtf(var + 1e-5f);
  h16* yp = y + (long long)row * C;
  for (int c = lane; c < C; c += 32)
    yp[c] = (h16)((xp[c] - mu) * inv * w[c] + bvec[c]);
}

// ---------------------------------------------------------------- WMMA GEMM
// C[M,N] = A[M,K] * Bw[N,K]^T (+bias) with templated epilogue.
// EPI 0: f16 store  1: gelu->f16  2: f32 residual add  3: f32 store
// Wave tile 32x64 (2x4 accumulators), workgroup tile 256x64.
// B tile (64 rows x 32 halves = 4KB) staged to LDS once per WG per K-step,
// double-buffered with async global->LDS copies.

template <int EPI>
__global__ __launch_bounds__(256, 1) void gemm_wmma_k(
    const h16* __restrict__ A, long long strideA,
    const h16* __restrict__ Bw,
    const float* __restrict__ bias,
    const float* __restrict__ resid,
    float* __restrict__ outF, h16* __restrict__ outH,
    int M, int N, int K) {
  __shared__ alignas(16) h16 bsh[2][64 * 32];

  const int lane = threadIdx.x & 31;
  const int wv   = threadIdx.x >> 5;
  const int m0   = blockIdx.y * 256 + wv * 32;
  const int n0   = blockIdx.x * 64;
  const int r    = lane & 15;
  const int ksel = (lane >> 4) << 3;

  int arow0 = m0 + r;      if (arow0 > M - 1) arow0 = M - 1;
  int arow1 = m0 + 16 + r; if (arow1 > M - 1) arow1 = M - 1;
  const h16* Ap0 = A + (long long)arow0 * strideA;
  const h16* Ap1 = A + (long long)arow1 * strideA;

  // staging role: thread copies 16B of the B tile
  const int trow = threadIdx.x >> 2;        // 0..63
  const int tseg = (threadIdx.x & 3) << 3;  // 0,8,16,24 halves
  int brow = n0 + trow; if (brow > N - 1) brow = N - 1;
  const h16* bsrc = Bw + (long long)brow * K + tseg;

  v8f acc0[4] = {};
  v8f acc1[4] = {};

  const int nk = K >> 5;
  stage16(bsrc, &bsh[0][trow * 32 + tseg]);

#pragma unroll 1
  for (int kt = 0; kt < nk; ++kt) {
    const int cur = kt & 1;
    if (kt + 1 < nk) {
      stage16(bsrc + (long long)(kt + 1) * 32, &bsh[cur ^ 1][trow * 32 + tseg]);
      stage_wait_one();
    } else {
      stage_wait_zero();
    }
    __syncthreads();  // B tile kt visible to all waves

    const int k0 = kt << 5;
    v16h af0 = ld_frag(Ap0 + k0 + ksel, Ap0 + k0 + 16 + ksel);
    v16h af1 = ld_frag(Ap1 + k0 + ksel, Ap1 + k0 + 16 + ksel);
#pragma unroll
    for (int j = 0; j < 4; ++j) {
      const h16* bp = &bsh[cur][(j * 16 + r) * 32];
      v16h bf = ld_frag(bp + ksel, bp + 16 + ksel);
      acc0[j] = wmma32(af0, bf, acc0[j]);
      acc1[j] = wmma32(af1, bf, acc1[j]);
    }
    __syncthreads();  // all waves done with tile kt before it is overwritten
  }

  const int rb = (lane >> 4) << 3;
#pragma unroll
  for (int half = 0; half < 2; ++half) {
    const int rbase = m0 + half * 16 + rb;
#pragma unroll
    for (int j = 0; j < 4; ++j) {
      int col = n0 + j * 16 + r;
      if (col >= N) continue;
      float bb = bias ? bias[col] : 0.0f;
#pragma unroll
      for (int e = 0; e < 8; ++e) {
        int row = rbase + e;
        if (row >= M) continue;
        float v = (half ? acc1[j][e] : acc0[j][e]) + bb;
        long long idx = (long long)row * N + col;
        if (EPI == 0) {
          outH[idx] = (h16)v;
        } else if (EPI == 1) {
          float g = 0.5f * v * (1.0f + erff(v * 0.70710678118f));
          outH[idx] = (h16)g;
        } else if (EPI == 2) {
          outF[idx] = resid[idx] + v;
        } else {
          outF[idx] = v;
        }
      }
    }
  }
}

// ---------------------------------------------------------------- attention
// One workgroup per (batch, head). 8 waves, each wave owns q-row-tiles of 16.
// Scores strip per wave: 16 x 224 f32 in LDS. V staged transposed in LDS.

#define NT 197
#define NP 224   // padded K for the P*V GEMM (7 * 32)
#define DK 64

__global__ __launch_bounds__(256, 1) void attn_k(const h16* __restrict__ qkv,
                                                 h16* __restrict__ o) {
  extern __shared__ char smem[];
  float* strips = (float*)smem;                                    // 8*16*NP f32
  h16*   vT     = (h16*)(smem + 8 * 16 * NP * sizeof(float));      // DK*NP f16

  const int bh   = blockIdx.x;
  const int b    = bh / 12;
  const int hh   = bh % 12;
  const int tid  = threadIdx.x;
  const int lane = tid & 31;
  const int wv   = tid >> 5;

  const long long tokbase = (long long)b * NT;

  // stage V transposed: vT[d][s] = v[s][d]; zero-pad s in [197,224)
  for (int idx = tid; idx < DK * NP; idx += blockDim.x) {
    int d = idx / NP, s = idx % NP;
    h16 val = (h16)0.0f;
    if (s < NT) val = qkv[(tokbase + s) * 2304 + 1536 + hh * 64 + d];
    vT[idx] = val;
  }
  __syncthreads();

  const int r    = lane & 15;
  const int ksel = (lane >> 4) << 3;
  const float scale = 0.125f;  // dk=64 -> 1/sqrt(64)

  for (int qt = wv; qt < 13; qt += 8) {
    float* P = strips + wv * 16 * NP;
    int qm = qt * 16 + r; if (qm > NT - 1) qm = NT - 1;
    const h16* qp = qkv + (tokbase + qm) * 2304 + hh * 64;

    // Q fragments hoisted: loaded once per q-tile.
    v16h qf0 = ld_frag(qp + ksel, qp + 16 + ksel);
    v16h qf1 = ld_frag(qp + 32 + ksel, qp + 48 + ksel);
    const int rb = (lane >> 4) << 3;

    // ---- S = scale * q k^T  (B columns = contiguous K rows)
#pragma unroll 1
    for (int jt = 0; jt < 13; ++jt) {
      int kn = jt * 16 + r; if (kn > NT - 1) kn = NT - 1;
      const h16* kp = qkv + (tokbase + kn) * 2304 + 768 + hh * 64;
      v16h kf0 = ld_frag(kp + ksel, kp + 16 + ksel);
      v16h kf1 = ld_frag(kp + 32 + ksel, kp + 48 + ksel);
      v8f s = {};
      s = wmma32(qf0, kf0, s);
      s = wmma32(qf1, kf1, s);
#pragma unroll
      for (int e = 0; e < 8; ++e)
        P[(rb + e) * NP + jt * 16 + r] = s[e] * scale;
    }

    // ---- row-wise softmax (wave cooperative; cols >= 197 zeroed)
#pragma unroll 1
    for (int rr = 0; rr < 16; ++rr) {
      float* row = P + rr * NP;
      float mx = -1e30f;
      for (int c = lane; c < NT; c += 32) mx = fmaxf(mx, row[c]);
      mx = wave_max(mx);
      float sm = 0.f;
      for (int c = lane; c < NT; c += 32) {
        float e = __expf(row[c] - mx);
        row[c] = e;
        sm += e;
      }
      sm = wave_sum(sm);
      float inv = 1.0f / sm;
      for (int c = lane; c < NT; c += 32) row[c] *= inv;
      for (int c = NT + lane; c < NP; c += 32) row[c] = 0.f;
    }

    // ---- O = P * V  (A frags converted f32->f16 from LDS strip)
    v8f acc[4] = {};
#pragma unroll 1
    for (int k0 = 0; k0 < NP; k0 += 32) {
      const float* prow = P + r * NP + k0;
      v16h af;
#pragma unroll
      for (int i = 0; i < 8; ++i) {
        af[i]     = (h16)prow[ksel + i];
        af[i + 8] = (h16)prow[16 + ksel + i];
      }
#pragma unroll
      for (int j = 0; j < 4; ++j) {
        const h16* bp = vT + (j * 16 + r) * NP + k0;
        v16h bf = ld_frag(bp + ksel, bp + 16 + ksel);
        acc[j] = wmma32(af, bf, acc[j]);
      }
    }
#pragma unroll
    for (int j = 0; j < 4; ++j) {
#pragma unroll
      for (int e = 0; e < 8; ++e) {
        int qrow = qt * 16 + rb + e;
        if (qrow < NT)
          o[(tokbase + qrow) * 768 + hh * 64 + j * 16 + r] = (h16)acc[j][e];
      }
    }
  }
}

// ---------------------------------------------------------------- host

extern "C" void kernel_launch(void* const* d_in, const int* in_sizes, int n_in,
                              void* d_out, int out_size, void* d_ws, size_t ws_size,
                              hipStream_t stream) {
  (void)in_sizes; (void)n_in; (void)out_size; (void)ws_size;
  const float* x      = (const float*)d_in[0];
  const float* cls    = (const float*)d_in[1];
  const float* qkv_w  = (const float*)d_in[2];
  const float* proj_w = (const float*)d_in[3];
  const float* proj_b = (const float*)d_in[4];
  const float* ln1_w  = (const float*)d_in[5];
  const float* ln1_b  = (const float*)d_in[6];
  const float* ln2_w  = (const float*)d_in[7];
  const float* ln2_b  = (const float*)d_in[8];
  const float* fc1_w  = (const float*)d_in[9];
  const float* fc1_b  = (const float*)d_in[10];
  const float* fc2_w  = (const float*)d_in[11];
  const float* fc2_b  = (const float*)d_in[12];
  const float* norm_w = (const float*)d_in[13];
  const float* norm_b = (const float*)d_in[14];
  const float* head_w = (const float*)d_in[15];
  const float* head_b = (const float*)d_in[16];
  float* out = (float*)d_out;

  const int Bc = 64, Ntok = 197, C = 768, D = 12, F = 3072, V = 1000;
  const long long M = (long long)Bc * Ntok;  // 12608

  char* ws = (char*)d_ws;
  size_t off = 0;
  auto alloc = [&](size_t bytes) -> void* {
    void* p = ws + off;
    off += (bytes + 255) & ~(size_t)255;
    return p;
  };
  float* h     = (float*)alloc((size_t)M * C * 4);
  h16* y       = (h16*)alloc((size_t)M * C * 2);
  h16* qkv     = (h16*)alloc((size_t)M * 3 * C * 2);
  h16* oatt    = (h16*)alloc((size_t)M * C * 2);
  h16* mbuf    = (h16*)alloc((size_t)M * F * 2);
  h16* qkvw_h  = (h16*)alloc((size_t)D * 3 * C * C * 2);
  h16* projw_h = (h16*)alloc((size_t)D * C * C * 2);
  h16* fc1w_h  = (h16*)alloc((size_t)D * F * C * 2);
  h16* fc2w_h  = (h16*)alloc((size_t)D * C * F * 2);
  h16* headw_h = (h16*)alloc((size_t)V * C * 2);

  auto cvt = [&](const float* src, h16* dst, long long n) {
    long long blocks = (n + 255) / 256;
    if (blocks > 16384) blocks = 16384;
    cvt_f16_k<<<(int)blocks, 256, 0, stream>>>(src, dst, n);
  };
  cvt(qkv_w, qkvw_h, (long long)D * 3 * C * C);
  cvt(proj_w, projw_h, (long long)D * C * C);
  cvt(fc1_w, fc1w_h, (long long)D * F * C);
  cvt(fc2_w, fc2w_h, (long long)D * C * F);
  cvt(head_w, headw_h, (long long)V * C);

  {
    long long total = M * C;
    embed_k<<<(int)((total + 255) / 256), 256, 0, stream>>>(x, cls, h);
  }

  auto ln = [&](const float* in, const float* w, const float* bb, h16* yo) {
    layernorm_k<<<(int)((M + 7) / 8), 256, 0, stream>>>(in, w, bb, yo, (int)M, C);
  };
  auto gemm = [&](int epi, const h16* A, long long sA, const h16* Bw,
                  const float* bias, const float* resid, float* oF, h16* oH,
                  int Mm, int Nn, int Kk) {
    dim3 grid((Nn + 63) / 64, (Mm + 255) / 256);
    switch (epi) {
      case 0: gemm_wmma_k<0><<<grid, 256, 0, stream>>>(A, sA, Bw, bias, resid, oF, oH, Mm, Nn, Kk); break;
      case 1: gemm_wmma_k<1><<<grid, 256, 0, stream>>>(A, sA, Bw, bias, resid, oF, oH, Mm, Nn, Kk); break;
      case 2: gemm_wmma_k<2><<<grid, 256, 0, stream>>>(A, sA, Bw, bias, resid, oF, oH, Mm, Nn, Kk); break;
      default: gemm_wmma_k<3><<<grid, 256, 0, stream>>>(A, sA, Bw, bias, resid, oF, oH, Mm, Nn, Kk); break;
    }
  };

  const size_t SMEM_ATTN = 8 * 16 * NP * sizeof(float) + DK * NP * sizeof(h16);

  for (int l = 0; l < D; ++l) {
    ln(h, ln1_w + (size_t)l * C, ln1_b + (size_t)l * C, y);
    gemm(0, y, C, qkvw_h + (size_t)l * 3 * C * C, nullptr, nullptr, nullptr, qkv,
         (int)M, 3 * C, C);
    attn_k<<<Bc * 12, 256, SMEM_ATTN, stream>>>(qkv, oatt);
    gemm(2, oatt, C, projw_h + (size_t)l * C * C, proj_b + (size_t)l * C, h, h,
         nullptr, (int)M, C, C);
    ln(h, ln2_w + (size_t)l * C, ln2_b + (size_t)l * C, y);
    gemm(1, y, C, fc1w_h + (size_t)l * F * C, fc1_b + (size_t)l * F, nullptr,
         nullptr, mbuf, (int)M, F, C);
    gemm(2, mbuf, F, fc2w_h + (size_t)l * C * F, fc2_b + (size_t)l * C, h, h,
         nullptr, (int)M, C, F);
  }

  ln(h, norm_w, norm_b, y);
  // head: A rows are the cls tokens -> row stride 197*C into y
  gemm(3, y, (long long)Ntok * C, headw_h, head_b, nullptr, out, nullptr, Bc, V, C);
}
